// WRGCN_2370821947940
// MI455X (gfx1250) — matrix operand order repaired and checked by reading, not compiled, not run on hardware
//
#include <hip/hip_runtime.h>
#include <math.h>

#define NN   100000
#define EE   3200000
#define RREL 8
#define FIN  128
#define DIMS 64
#define CCLS 40

typedef __attribute__((ext_vector_type(16))) __bf16 v16bf;
typedef __attribute__((ext_vector_type(8)))  __bf16 v8bf;
typedef __attribute__((ext_vector_type(8)))  float  v8f;

// ---------------- f32 -> bf16 conversion ----------------
__global__ void k_f32_to_bf16(const float* __restrict__ in, __bf16* __restrict__ out, int n) {
  int i = blockIdx.x * blockDim.x + threadIdx.x;
  if (i < n) out[i] = (__bf16)in[i];
}

// ---------------- weight repack into WMMA B-register layout ----------------
// Blob = 512 bf16 (1KB) per (rel, ktile, ntile). Lane l holds 16 bf16:
//   element e -> B[K = kt*32 + (l/16)*16 + e][N = nt*16 + (l%16)]  (zero-pad cols >= OUT)
template<int KSTEPS, int NTILES, int OUT>
__global__ void k_repack(const float* __restrict__ Wrel, const float* __restrict__ Wroot,
                         __bf16* __restrict__ pack) {
  int t = blockIdx.x * blockDim.x + threadIdx.x;
  const int total = (RREL + 1) * KSTEPS * NTILES * 32;
  if (t >= total) return;
  int lane = t & 31;
  int blob = t >> 5;
  int nt = blob % NTILES;
  int kt = (blob / NTILES) % KSTEPS;
  int rr = blob / (NTILES * KSTEPS);
  const int K = KSTEPS * 32;
  const float* W = (rr < RREL) ? (Wrel + (size_t)rr * K * OUT) : Wroot;
  int col   = nt * 16 + (lane & 15);
  int krow0 = kt * 32 + (lane >> 4) * 16;
  __bf16* dst = pack + (size_t)blob * 512 + lane * 16;
#pragma unroll
  for (int e = 0; e < 16; ++e) {
    float v = (col < OUT) ? W[(size_t)(krow0 + e) * OUT + col] : 0.0f;
    dst[e] = (__bf16)v;
  }
}

// ---------------- WMMA GEMM: Out[rr] = A (nrows x K, bf16) @ Wpack[rr] ----------------
// One wave computes a 16-row slab x (NTILES*16) cols with v_wmma_f32_16x16x32_bf16.
// blockIdx.z = relation (RREL == root transform).
template<int KSTEPS, int NTILES, int OUTCOLS>
__global__ void k_wmma_gemm(const __bf16* __restrict__ A, const __bf16* __restrict__ Wpack,
                            float* __restrict__ Out, int nrows) {
  const int LDA  = KSTEPS * 32;
  int lane = threadIdx.x & 31;
  int wave = threadIdx.x >> 5;
  int row0 = (blockIdx.x * 8 + wave) * 16;
  if (row0 >= nrows) return;                 // wave-uniform exit (EXEC all-1s for WMMA)
  int rr    = blockIdx.z;
  int m     = lane & 15;                     // A row within tile (ISA 16-bit A layout)
  int khalf = lane >> 4;                     // which K half-group this lane holds
  int rowA  = row0 + m;
  if (rowA >= nrows) rowA = nrows - 1;       // clamp (stores are guarded)
  const __bf16* arow  = A + (size_t)rowA * LDA;
  const __bf16* wbase = Wpack + (size_t)rr * KSTEPS * NTILES * 512;

  v8f acc[NTILES];
#pragma unroll
  for (int nt = 0; nt < NTILES; ++nt)
#pragma unroll
    for (int j = 0; j < 8; ++j) acc[nt][j] = 0.0f;

#pragma unroll
  for (int kt = 0; kt < KSTEPS; ++kt) {
    int kbase = kt * 32;
    // A lanes: l<16 hold K[0..15]; l>=16 hold K[16..31]
    v8bf lo = *(const v8bf*)(arow + kbase + khalf * 16);
    v8bf hi = *(const v8bf*)(arow + kbase + khalf * 16 + 8);
    v16bf a;
#pragma unroll
    for (int i = 0; i < 8; ++i) { a[i] = lo[i]; a[i + 8] = hi[i]; }
#pragma unroll
    for (int nt = 0; nt < NTILES; ++nt) {
      v16bf b = *(const v16bf*)(wbase + ((size_t)kt * NTILES + nt) * 512 + lane * 16);
      acc[nt] = __builtin_amdgcn_wmma_f32_16x16x32_bf16(
          false, a, false, b, (short)0, acc[nt], false, false);
    }
  }

  // C/D layout: VGPR j, lane l -> M = j + (l>=16 ? 8 : 0), N = l%16
  float* obase = Out + (size_t)rr * nrows * OUTCOLS;
#pragma unroll
  for (int nt = 0; nt < NTILES; ++nt) {
    int col = nt * 16 + (lane & 15);
    if (col < OUTCOLS) {
#pragma unroll
      for (int j = 0; j < 8; ++j) {
        int orow = row0 + j + (lane >> 4) * 8;
        if (orow < nrows) obase[(size_t)orow * OUTCOLS + col] = acc[nt][j];
      }
    }
  }
}

// ---------------- layer-1 edge gather/scatter (64 f32 per edge, float4 chunks) ----------------
__global__ void k_edge_l1(const int* __restrict__ src, const int* __restrict__ dst,
                          const float* __restrict__ ew, const int* __restrict__ ecol,
                          const float* __restrict__ h1, float* __restrict__ agg) {
  long long idx = (long long)blockIdx.x * blockDim.x + threadIdx.x;
  if (idx >= (long long)EE * 16) return;
  int e = (int)(idx >> 4);
  int c = (int)(idx & 15) * 4;
  int s = src[e], d = dst[e], r = ecol[e];
  float w = ew[e];
  float4 v = *reinterpret_cast<const float4*>(h1 + ((size_t)r * NN + s) * DIMS + c);
  float* a = agg + (size_t)d * DIMS + c;
  atomicAdd(a + 0, v.x * w);
  atomicAdd(a + 1, v.y * w);
  atomicAdd(a + 2, v.z * w);
  atomicAdd(a + 3, v.w * w);
}

// ---------------- layer-2 edge gather/scatter (40 f32 per edge, 10 float4 chunks) ----------------
__global__ void k_edge_l2(const int* __restrict__ src, const int* __restrict__ dst,
                          const float* __restrict__ ew, const int* __restrict__ ecol,
                          const float* __restrict__ h2, float* __restrict__ agg) {
  long long idx = (long long)blockIdx.x * blockDim.x + threadIdx.x;
  if (idx >= (long long)EE * 10) return;
  int e = (int)(idx / 10);
  int c = (int)(idx - (long long)e * 10) * 4;
  int s = src[e], d = dst[e], r = ecol[e];
  float w = ew[e];
  float4 v = *reinterpret_cast<const float4*>(h2 + ((size_t)r * NN + s) * CCLS + c);
  float* a = agg + (size_t)d * CCLS + c;
  atomicAdd(a + 0, v.x * w);
  atomicAdd(a + 1, v.y * w);
  atomicAdd(a + 2, v.z * w);
  atomicAdd(a + 3, v.w * w);
}

// ---------------- h = relu(agg1 + root1 + b1), converted to bf16 for layer-2 WMMA ----------------
__global__ void k_relu_combine(const float* __restrict__ agg, const float* __restrict__ root,
                               const float* __restrict__ b, __bf16* __restrict__ hb) {
  int i = blockIdx.x * blockDim.x + threadIdx.x;
  if (i >= NN * DIMS) return;
  int c = i & (DIMS - 1);
  float v = agg[i] + root[i] + b[c];
  hb[i] = (__bf16)fmaxf(v, 0.0f);
}

// ---------------- out = agg2 + root2 + b2 ; write (log_softmax, raw) ; wave32 per node ----------------
__global__ void k_final(const float* __restrict__ agg, const float* __restrict__ root,
                        const float* __restrict__ b, float* __restrict__ out) {
  int lane = threadIdx.x & 31;
  int node = blockIdx.x * 8 + (threadIdx.x >> 5);
  if (node >= NN) return;
  const float* a = agg + (size_t)node * CCLS;
  const float* r = root + (size_t)node * CCLS;
  float v1 = a[lane] + r[lane] + b[lane];                        // lane < 32 < 40
  bool hi = lane < (CCLS - 32);
  float v2 = hi ? (a[lane + 32] + r[lane + 32] + b[lane + 32]) : -__builtin_inff();
  float m = fmaxf(v1, v2);
#pragma unroll
  for (int o = 16; o >= 1; o >>= 1) m = fmaxf(m, __shfl_xor(m, o, 32));
  float s = __expf(v1 - m) + (hi ? __expf(v2 - m) : 0.0f);
#pragma unroll
  for (int o = 16; o >= 1; o >>= 1) s += __shfl_xor(s, o, 32);
  float ls = __logf(s);
  float* lsm = out;
  float* raw = out + (size_t)NN * CCLS;
  lsm[(size_t)node * CCLS + lane] = v1 - m - ls;
  raw[(size_t)node * CCLS + lane] = v1;
  if (hi) {
    lsm[(size_t)node * CCLS + lane + 32] = v2 - m - ls;
    raw[(size_t)node * CCLS + lane + 32] = v2;
  }
}

extern "C" void kernel_launch(void* const* d_in, const int* in_sizes, int n_in,
                              void* d_out, int out_size, void* d_ws, size_t ws_size,
                              hipStream_t stream) {
  (void)in_sizes; (void)n_in; (void)out_size; (void)ws_size;
  const float* x      = (const float*)d_in[0];
  const int*   ei     = (const int*)  d_in[1];
  const float* ew     = (const float*)d_in[2];
  const int*   ecol   = (const int*)  d_in[3];
  const float* W1rel  = (const float*)d_in[4];
  const float* W1root = (const float*)d_in[5];
  const float* b1     = (const float*)d_in[6];
  const float* W2rel  = (const float*)d_in[7];
  const float* W2root = (const float*)d_in[8];
  const float* b2     = (const float*)d_in[9];
  const int* src = ei;
  const int* dst = ei + EE;

  char* ws = (char*)d_ws;
  size_t off = 0;
  auto alloc = [&](size_t bytes) -> void* {
    void* p = ws + off;
    off += (bytes + 255) & ~(size_t)255;
    return p;
  };
  float*  h1   = (float*) alloc((size_t)(RREL + 1) * NN * DIMS * 4);  // 8 rel + root slabs
  float*  h2   = (float*) alloc((size_t)(RREL + 1) * NN * CCLS * 4);
  float*  agg1 = (float*) alloc((size_t)NN * DIMS * 4);
  float*  agg2 = (float*) alloc((size_t)NN * CCLS * 4);
  __bf16* xb   = (__bf16*)alloc((size_t)NN * FIN * 2);
  __bf16* hb   = (__bf16*)alloc((size_t)NN * DIMS * 2);
  __bf16* wp1  = (__bf16*)alloc((size_t)(RREL + 1) * 4 * 4 * 1024);
  __bf16* wp2  = (__bf16*)alloc((size_t)(RREL + 1) * 2 * 3 * 1024);
  float* root1 = h1 + (size_t)RREL * NN * DIMS;
  float* root2 = h2 + (size_t)RREL * NN * CCLS;

  hipMemsetAsync(agg1, 0, (size_t)NN * DIMS * 4, stream);
  hipMemsetAsync(agg2, 0, (size_t)NN * CCLS * 4, stream);

  // Stage bf16 activations + repack weights into WMMA B layout.
  k_f32_to_bf16<<<(NN * FIN + 255) / 256, 256, 0, stream>>>(x, xb, NN * FIN);
  k_repack<4, 4, DIMS><<<((RREL + 1) * 4 * 4 * 32 + 255) / 256, 256, 0, stream>>>(W1rel, W1root, wp1);
  k_repack<2, 3, CCLS><<<((RREL + 1) * 2 * 3 * 32 + 255) / 256, 256, 0, stream>>>(W2rel, W2root, wp2);

  // Layer 1: h1[r] = x @ W1_rel[r] (r=0..7), root1 = x @ W1_root   (K=128, 64 cols)
  dim3 g1((NN + 127) / 128, 1, RREL + 1);
  k_wmma_gemm<4, 4, DIMS><<<g1, 256, 0, stream>>>(xb, wp1, h1, NN);

  long long t1 = (long long)EE * 16;
  k_edge_l1<<<(unsigned)((t1 + 255) / 256), 256, 0, stream>>>(src, dst, ew, ecol, h1, agg1);

  k_relu_combine<<<(NN * DIMS + 255) / 256, 256, 0, stream>>>(agg1, root1, b1, hb);

  // Layer 2: h2[r] = h @ W2_rel[r], root2 = h @ W2_root   (K=64, 40 cols -> 48 computed)
  dim3 g2((NN + 127) / 128, 1, RREL + 1);
  k_wmma_gemm<2, 3, CCLS><<<g2, 256, 0, stream>>>(hb, wp2, h2, NN);

  long long t2 = (long long)EE * 10;
  k_edge_l2<<<(unsigned)((t2 + 255) / 256), 256, 0, stream>>>(src, dst, ew, ecol, h2, agg2);

  k_final<<<(NN + 7) / 8, 256, 0, stream>>>(agg2, root2, b2, (float*)d_out);
}
